// FuturesModel_59614146068528
// MI455X (gfx1250) — matrix operand order repaired and checked by reading, not compiled, not run on hardware
//
#include <hip/hip_runtime.h>

// MI455X / gfx1250 implementation.
// - All dense contractions use v_wmma_f32_16x16x32_f16 (f16 in, f32 accum).
// - Weights converted once per call to f16 transposed [N][K] (contiguous
//   B-fragments); activations kept in an f16 mirror by their producers.
// - GEMM stages the block-shared 16xK A-tile into LDS via
//   GLOBAL_LOAD_ASYNC_TO_LDS_B128 (+ s_wait_asynccnt) when available.
// Workspace requirement: ~45 MB.

typedef __attribute__((ext_vector_type(16))) _Float16 v16h;
typedef __attribute__((ext_vector_type(8)))  _Float16 v8h;
typedef __attribute__((ext_vector_type(8)))  float    v8f;

#define NTOK 2048   // B*S = 16*128
#define EPS_ 1e-5f

#if __has_builtin(__builtin_amdgcn_global_load_async_to_lds_b128) && \
    __has_builtin(__builtin_amdgcn_s_wait_asynccnt)
#define USE_ASYNC_COPY 1
#else
#define USE_ASYNC_COPY 0
#endif

#if USE_ASYNC_COPY
typedef int v4ia __attribute__((vector_size(16)));
typedef __attribute__((address_space(1))) v4ia* gp1_t;
typedef __attribute__((address_space(3))) v4ia* lp3_t;
#endif

__device__ __forceinline__ void copy16_g2lds(const _Float16* g, _Float16* l) {
#if USE_ASYNC_COPY
  __builtin_amdgcn_global_load_async_to_lds_b128((gp1_t)g, (lp3_t)l, 0, 0);
#else
  *(v8h*)l = *(const v8h*)g;
#endif
}

__device__ __forceinline__ void async_copy_fence() {
#if USE_ASYNC_COPY
  __builtin_amdgcn_s_wait_asynccnt(0);
#endif
}

__device__ __forceinline__ float gelu_exact(float x) {
  return 0.5f * x * (1.0f + erff(x * 0.70710678118654752440f));
}

__device__ __forceinline__ v8f wmma_f16(v16h a, v16h b, v8f c) {
  // D = A(16x32 f16) * B(32x16 f16) + C(16x16 f32)
  return __builtin_amdgcn_wmma_f32_16x16x32_f16(false, a, false, b, (short)0, c,
                                                false, false);
}

// ---------------------------------------------------------------------------
// Generic GEMM: C[M,N] = act(Ah[M,K](f16) @ Bt[N,K](f16)^T + bias)(*scale)(+=)
// Block = 8 waves sharing one 16-row A-tile staged in LDS (async copy);
// each wave computes one 16x16 tile along N.
// A-frag lane layout: m = m0+(l&15), k = k0 + g*16 + e
// B-frag lane layout: n = n0+(l&15), k = k0 + g*16 + e   (g = l>>4)
// ---------------------------------------------------------------------------
template <int ACT, bool ACCUM, bool SCALE, bool OUTH, int KT>
__global__ __launch_bounds__(256) void gemm_kernel(
    const _Float16* __restrict__ Ah, const _Float16* __restrict__ Bt,
    const float* __restrict__ bias, void* __restrict__ Cout, int M, int N,
    const float* __restrict__ rowScale, int scaleStride, int scaleIdx) {
  __shared__ _Float16 As[16][KT];
  const int m0 = blockIdx.y * 16;
  // cooperative 16xKT f16 A-tile load, 16-byte chunks
  for (int c = threadIdx.x; c < 2 * KT; c += 256) {
    const int idx = c * 8;               // half index
    const int r = idx / KT, col = idx % KT;
    copy16_g2lds(Ah + (size_t)(m0 + r) * KT + col, &As[r][col]);
  }
  async_copy_fence();
  __syncthreads();

  const int wave = threadIdx.x >> 5;
  const int l = threadIdx.x & 31;
  const int half = l >> 4;
  const int ln = l & 15;
  const int n0 = (blockIdx.x * 8 + wave) * 16;
  if (n0 >= N) return;

  const _Float16* arow = &As[ln][half * 16];
  const _Float16* brow = Bt + (size_t)(n0 + ln) * KT + half * 16;
  v8f acc = {0.f, 0.f, 0.f, 0.f, 0.f, 0.f, 0.f, 0.f};
#pragma unroll
  for (int k0 = 0; k0 < KT; k0 += 32) {
    v8h alo = *(const v8h*)(arow + k0);
    v8h ahi = *(const v8h*)(arow + k0 + 8);
    v16h afr;
#pragma unroll
    for (int j = 0; j < 8; ++j) { afr[j] = alo[j]; afr[8 + j] = ahi[j]; }
    v16h bfr = *(const v16h*)(brow + k0);
    if (k0 + 32 < KT) __builtin_prefetch((const void*)(brow + k0 + 32), 0, 1);
    acc = wmma_f16(afr, bfr, acc);
  }
#pragma unroll
  for (int v = 0; v < 8; ++v) {
    const int m = m0 + v + half * 8;   // C layout: vgpr v -> row v (+8 upper half)
    const int n = n0 + ln;
    float val = acc[v];
    if (bias) val += bias[n];
    if (ACT == 1) val = gelu_exact(val);
    if (SCALE) val *= rowScale[(size_t)m * scaleStride + scaleIdx];
    const size_t o = (size_t)m * N + n;
    if (OUTH) ((_Float16*)Cout)[o] = (_Float16)val;
    else if (ACCUM) ((float*)Cout)[o] += val;
    else ((float*)Cout)[o] = val;
  }
}

// ---------------------------------------------------------------------------
// Attention: one block per (batch, head). S=128, HD=32, full softmax.
// qkv arrives as f16. Stage 1: scores = Q K^T / sqrt(32) via WMMA; softmax in
// registers with 16-lane shfl reductions; probs -> LDS f16. Stage 2: P V.
// ---------------------------------------------------------------------------
__global__ __launch_bounds__(256) void attention_kernel(
    const _Float16* __restrict__ qkv, _Float16* __restrict__ o) {
  __shared__ _Float16 qs[128][32];
  __shared__ _Float16 ks[128][32];
  __shared__ _Float16 vsT[32][128];
  __shared__ _Float16 ph[128][128];
  const int bh = blockIdx.x;
  const int b = bh >> 3, h = bh & 7;
  const _Float16* base = qkv + (size_t)b * 128 * 768;
  for (int idx = threadIdx.x; idx < 128 * 32; idx += 256) {
    const int s = idx >> 5, d = idx & 31;
    qs[s][d] = base[s * 768 + h * 32 + d];
    ks[s][d] = base[s * 768 + 256 + h * 32 + d];
    vsT[d][s] = base[s * 768 + 512 + h * 32 + d];
  }
  __syncthreads();
  const int wave = threadIdx.x >> 5;
  const int l = threadIdx.x & 31;
  const int half = l >> 4, ln = l & 15;
  const int m0 = wave * 16;  // this wave owns query rows [m0, m0+16)
  v8f acc[8];
  {
    v16h afr;
    const _Float16* ap = &qs[m0 + ln][half * 16];
    v8h alo = *(const v8h*)ap;
    v8h ahi = *(const v8h*)(ap + 8);
#pragma unroll
    for (int j = 0; j < 8; ++j) { afr[j] = alo[j]; afr[8 + j] = ahi[j]; }
#pragma unroll
    for (int tn = 0; tn < 8; ++tn) {
      v16h bfr = *(const v16h*)&ks[tn * 16 + ln][half * 16];
      v8f z = {0.f, 0.f, 0.f, 0.f, 0.f, 0.f, 0.f, 0.f};
      acc[tn] = wmma_f16(afr, bfr, z);
    }
  }
  const float scl = 0.17677669529663687f;  // 1/sqrt(32)
#pragma unroll
  for (int v = 0; v < 8; ++v) {
    float pm = -3.0e38f;
#pragma unroll
    for (int tn = 0; tn < 8; ++tn) {
      acc[tn][v] *= scl;
      pm = fmaxf(pm, acc[tn][v]);
    }
    for (int m = 1; m < 16; m <<= 1) pm = fmaxf(pm, __shfl_xor(pm, m, 32));
    float e[8], ps = 0.f;
#pragma unroll
    for (int tn = 0; tn < 8; ++tn) { e[tn] = expf(acc[tn][v] - pm); ps += e[tn]; }
    for (int m = 1; m < 16; m <<= 1) ps += __shfl_xor(ps, m, 32);
    const float inv = 1.0f / ps;
    const int row = m0 + v + half * 8;
#pragma unroll
    for (int tn = 0; tn < 8; ++tn) ph[row][tn * 16 + ln] = (_Float16)(e[tn] * inv);
  }
  __syncthreads();
  v8f o0 = {0.f, 0.f, 0.f, 0.f, 0.f, 0.f, 0.f, 0.f};
  v8f o1 = {0.f, 0.f, 0.f, 0.f, 0.f, 0.f, 0.f, 0.f};
  for (int kk = 0; kk < 4; ++kk) {  // K = 128 keys, chunks of 32
    v16h afr;
    const _Float16* ap = &ph[m0 + ln][kk * 32 + half * 16];
    v8h alo = *(const v8h*)ap;
    v8h ahi = *(const v8h*)(ap + 8);
#pragma unroll
    for (int j = 0; j < 8; ++j) { afr[j] = alo[j]; afr[8 + j] = ahi[j]; }
    v16h b0 = *(const v16h*)&vsT[ln][kk * 32 + half * 16];
    v16h b1 = *(const v16h*)&vsT[16 + ln][kk * 32 + half * 16];
    o0 = wmma_f16(afr, b0, o0);
    o1 = wmma_f16(afr, b1, o1);
  }
#pragma unroll
  for (int v = 0; v < 8; ++v) {
    const int m = m0 + v + half * 8;
    _Float16* orow = o + (size_t)(b * 128 + m) * 256 + h * 32;
    orow[ln] = (_Float16)o0[v];
    orow[16 + ln] = (_Float16)o1[v];
  }
}

// x = LN(x + add); also writes f16 mirror xh (one block per token row)
__global__ __launch_bounds__(256) void residual_ln_kernel(
    float* __restrict__ x, _Float16* __restrict__ xh,
    const float* __restrict__ add, const float* __restrict__ g,
    const float* __restrict__ bb) {
  __shared__ float s1[8], s2[8];
  const int row = blockIdx.x, t = threadIdx.x;
  float v = x[(size_t)row * 256 + t];
  if (add) v += add[(size_t)row * 256 + t];
  float a = v, q = v * v;
  for (int m = 1; m < 32; m <<= 1) { a += __shfl_xor(a, m, 32); q += __shfl_xor(q, m, 32); }
  if ((t & 31) == 0) { s1[t >> 5] = a; s2[t >> 5] = q; }
  __syncthreads();
  float sum = 0.f, sq = 0.f;
#pragma unroll
  for (int i = 0; i < 8; ++i) { sum += s1[i]; sq += s2[i]; }
  const float mean = sum * (1.f / 256.f);
  const float var = sq * (1.f / 256.f) - mean * mean;
  const float inv = rsqrtf(var + EPS_);
  const float r = (v - mean) * inv * g[t] + bb[t];
  x[(size_t)row * 256 + t] = r;
  xh[(size_t)row * 256 + t] = (_Float16)r;
}

// x = LN3(gamma[b]*(x + ssm_out) + beta[b]); also writes f16 mirror
__global__ __launch_bounds__(256) void ssm_film_ln_kernel(
    float* __restrict__ x, _Float16* __restrict__ xh,
    const float* __restrict__ tmp, const float* __restrict__ gamma,
    const float* __restrict__ beta, const float* __restrict__ g,
    const float* __restrict__ bb) {
  __shared__ float s1[8], s2[8];
  const int row = blockIdx.x, t = threadIdx.x;
  const int b = row >> 7;
  float v = x[(size_t)row * 256 + t] + tmp[(size_t)row * 256 + t];
  v = gamma[b * 256 + t] * v + beta[b * 256 + t];
  float a = v, q = v * v;
  for (int m = 1; m < 32; m <<= 1) { a += __shfl_xor(a, m, 32); q += __shfl_xor(q, m, 32); }
  if ((t & 31) == 0) { s1[t >> 5] = a; s2[t >> 5] = q; }
  __syncthreads();
  float sum = 0.f, sq = 0.f;
#pragma unroll
  for (int i = 0; i < 8; ++i) { sum += s1[i]; sq += s2[i]; }
  const float mean = sum * (1.f / 256.f);
  const float var = sq * (1.f / 256.f) - mean * mean;
  const float inv = rsqrtf(var + EPS_);
  const float r = (v - mean) * inv * g[t] + bb[t];
  x[(size_t)row * 256 + t] = r;
  xh[(size_t)row * 256 + t] = (_Float16)r;
}

// gate softmax (E=8) + entropy / std(ddof=1) accumulation. One thread per row.
__global__ __launch_bounds__(256) void gate_kernel(
    const float* __restrict__ x, const float* __restrict__ gw,
    const float* __restrict__ gb, float* __restrict__ gate,
    float* __restrict__ entAcc, float* __restrict__ stdAcc) {
  const int row = blockIdx.x * blockDim.x + threadIdx.x;
  float lg[8];
#pragma unroll
  for (int e = 0; e < 8; ++e) lg[e] = gb[e];
  const float* xr = x + (size_t)row * 256;
  for (int d = 0; d < 256; ++d) {
    const float xv = xr[d];
    const float* gr = gw + d * 8;
#pragma unroll
    for (int e = 0; e < 8; ++e) lg[e] += xv * gr[e];
  }
  float mx = lg[0];
#pragma unroll
  for (int e = 1; e < 8; ++e) mx = fmaxf(mx, lg[e]);
  float s = 0.f;
#pragma unroll
  for (int e = 0; e < 8; ++e) { lg[e] = expf(lg[e] - mx); s += lg[e]; }
  const float inv = 1.0f / s;
  float ent = 0.f, mu = 0.f;
#pragma unroll
  for (int e = 0; e < 8; ++e) {
    lg[e] *= inv;
    gate[(size_t)row * 8 + e] = lg[e];
    ent -= lg[e] * logf(lg[e] + 1e-10f);
    mu += lg[e];
  }
  mu *= 0.125f;
  float var = 0.f;
#pragma unroll
  for (int e = 0; e < 8; ++e) { const float d2 = lg[e] - mu; var += d2 * d2; }
  var *= (1.f / 7.f);
  atomicAdd(entAcc, ent);
  atomicAdd(stdAcc, sqrtf(var));
}

// SSM recurrence h_t = h_{t-1} A^T + xB_t. One block per batch, 64 threads.
// Writes hs as f16 (consumed only as a GEMM A-operand).
__global__ void ssm_kernel(const float* __restrict__ A,
                           const float* __restrict__ h0,
                           const float* __restrict__ xB,
                           _Float16* __restrict__ hsH) {
  __shared__ float At[64 * 64];  // At[m][n] = A[n][m] (conflict-free reads)
  __shared__ float sh[64];
  const int b = blockIdx.x, t = threadIdx.x;
  for (int m = 0; m < 64; ++m) At[m * 64 + t] = A[t * 64 + m];
  sh[t] = h0[t];
  __syncthreads();
  for (int s = 0; s < 128; ++s) {
    float acc = xB[((size_t)(b * 128 + s)) * 64 + t];
    for (int m = 0; m < 64; ++m) acc += sh[m] * At[m * 64 + t];
    __syncthreads();
    sh[t] = acc;
    hsH[((size_t)(b * 128 + s)) * 64 + t] = (_Float16)acc;
    __syncthreads();
  }
}

__global__ void film_kernel(const float* __restrict__ aw,
                            const float* __restrict__ gw,
                            const float* __restrict__ gb,
                            const float* __restrict__ bw,
                            const float* __restrict__ bbv,
                            float* __restrict__ gamma, float* __restrict__ beta) {
  const int b = blockIdx.x, d = threadIdx.x;
  float g = gb[d], bt = bbv[d];
#pragma unroll
  for (int a = 0; a < 12; ++a) {
    const float w = aw[b * 12 + a];
    g += w * gw[a * 256 + d];
    bt += w * bw[a * 256 + d];
  }
  gamma[b * 256 + d] = g;
  beta[b * 256 + d] = bt;
}

__global__ void pool_kernel(const float* __restrict__ x,
                            const int* __restrict__ tokens,
                            float* __restrict__ pooled) {
  const int b = blockIdx.x, d = threadIdx.x;
  float sum = 0.f;
  int cnt = 0;
  for (int s = 0; s < 128; ++s) {
    if (tokens[b * 128 + s] != 0) {
      sum += x[((size_t)(b * 128 + s)) * 256 + d];
      ++cnt;
    }
  }
  pooled[b * 256 + d] = sum / fmaxf((float)cnt, 1.0f);
}

__global__ void ah1_kernel(const float* __restrict__ pooled,
                           const float* __restrict__ w1,
                           const float* __restrict__ b1, float* __restrict__ tb) {
  const int b = blockIdx.x, j = threadIdx.x;
  float acc = b1[j];
  for (int d = 0; d < 256; ++d) acc += pooled[b * 256 + d] * w1[d * 256 + j];
  tb[b * 256 + j] = gelu_exact(acc);
}

__global__ void ah2_kernel(const float* __restrict__ tb,
                           const float* __restrict__ w2,
                           const float* __restrict__ b2, float* __restrict__ out) {
  const int b = blockIdx.x, e = threadIdx.x;
  if (e >= 12) return;
  float acc = b2[e];
  for (int d = 0; d < 256; ++d) acc += tb[b * 256 + d] * w2[d * 12 + e];
  out[b * 12 + e] = acc;
}

__global__ void embed_kernel(const int* __restrict__ tokens,
                             const float* __restrict__ te,
                             const float* __restrict__ pe,
                             float* __restrict__ x, _Float16* __restrict__ xh) {
  const int row = blockIdx.x, d = threadIdx.x;
  const int s = row & 127;
  const int tok = tokens[row];
  const float v = te[(size_t)tok * 256 + d] + pe[s * 256 + d];
  x[(size_t)row * 256 + d] = v;
  xh[(size_t)row * 256 + d] = (_Float16)v;
}

// dst[n*K+k] (f16) = src[k*N+n] (f32); batched via blockIdx.z
__global__ void cvtT_kernel(const float* __restrict__ src,
                            _Float16* __restrict__ dst, int K, int N) {
  const size_t z = blockIdx.z;
  src += z * (size_t)K * N;
  dst += z * (size_t)N * K;
  const int idx = blockIdx.x * 256 + threadIdx.x;
  if (idx >= N * K) return;
  const int n = idx / K, k = idx - n * K;
  dst[idx] = (_Float16)src[(size_t)k * N + n];
}

__global__ void cvtC_kernel(const float* __restrict__ src,
                            _Float16* __restrict__ dst, int count) {
  const int idx = blockIdx.x * 256 + threadIdx.x;
  if (idx < count) dst[idx] = (_Float16)src[idx];
}

__global__ void init_kernel(float* stats) {
  if (threadIdx.x == 0) { stats[0] = 0.f; stats[1] = 0.f; }
}

__global__ void fin_kernel(const float* stats, float* out_ent, float* out_std) {
  if (threadIdx.x == 0) {
    const float inv = 1.0f / 8192.0f;  // L*B*S
    *out_ent = stats[0] * inv;
    *out_std = stats[1] * inv;
  }
}

extern "C" void kernel_launch(void* const* d_in, const int* in_sizes, int n_in,
                              void* d_out, int out_size, void* d_ws, size_t ws_size,
                              hipStream_t stream) {
  (void)in_sizes; (void)n_in; (void)out_size; (void)ws_size;
  const int* tokens = (const int*)d_in[0];
  const float* axw = (const float*)d_in[1];
  const float* tok_emb = (const float*)d_in[2];
  const float* pos_emb = (const float*)d_in[3];
  const float* attn_in_w = (const float*)d_in[4];
  const float* attn_in_b = (const float*)d_in[5];
  const float* attn_out_w = (const float*)d_in[6];
  const float* attn_out_b = (const float*)d_in[7];
  const float* gate_w = (const float*)d_in[8];
  const float* gate_b = (const float*)d_in[9];
  const float* moe_w1 = (const float*)d_in[10];
  const float* moe_b1 = (const float*)d_in[11];
  const float* moe_w2 = (const float*)d_in[12];
  const float* moe_b2 = (const float*)d_in[13];
  const float* ssm_A = (const float*)d_in[14];
  const float* ssm_Bm = (const float*)d_in[15];
  const float* ssm_Cm = (const float*)d_in[16];
  const float* ssm_Dm = (const float*)d_in[17];
  const float* ssm_h0 = (const float*)d_in[18];
  const float* film_gw = (const float*)d_in[19];
  const float* film_gb = (const float*)d_in[20];
  const float* film_bw = (const float*)d_in[21];
  const float* film_bb = (const float*)d_in[22];
  const float* ln1_g = (const float*)d_in[23];
  const float* ln1_b = (const float*)d_in[24];
  const float* ln2_g = (const float*)d_in[25];
  const float* ln2_b = (const float*)d_in[26];
  const float* ln3_g = (const float*)d_in[27];
  const float* ln3_b = (const float*)d_in[28];
  const float* ah_w1 = (const float*)d_in[29];
  const float* ah_b1 = (const float*)d_in[30];
  const float* ah_w2 = (const float*)d_in[31];
  const float* ah_b2 = (const float*)d_in[32];
  const float* lm_w = (const float*)d_in[33];
  const float* lm_b = (const float*)d_in[34];
  float* out = (float*)d_out;

  char* wsb = (char*)d_ws;
  size_t off = 0;
  auto alloc = [&](size_t bytes) -> char* {
    char* p = wsb + off;
    off = (off + bytes + 255) & ~(size_t)255;
    return p;
  };
  float* stats = (float*)alloc(2 * sizeof(float));
  float* x = (float*)alloc((size_t)NTOK * 256 * 4);
  _Float16* xh = (_Float16*)alloc((size_t)NTOK * 256 * 2);
  _Float16* qkvH = (_Float16*)alloc((size_t)NTOK * 768 * 2);
  _Float16* aBufH = (_Float16*)alloc((size_t)NTOK * 256 * 2);
  float* pBuf = (float*)alloc((size_t)NTOK * 256 * 4);
  float* gateBuf = (float*)alloc((size_t)NTOK * 8 * 4);
  _Float16* hBufH = (_Float16*)alloc((size_t)NTOK * 256 * 2);
  float* moeBuf = (float*)alloc((size_t)NTOK * 256 * 4);
  float* xBBuf = (float*)alloc((size_t)NTOK * 64 * 4);
  _Float16* hsH = (_Float16*)alloc((size_t)NTOK * 64 * 2);
  float* tmpBuf = (float*)alloc((size_t)NTOK * 256 * 4);
  float* gammaB = (float*)alloc(16 * 256 * 4);
  float* betaB = (float*)alloc(16 * 256 * 4);
  float* pooledB = (float*)alloc(16 * 256 * 4);
  float* tB = (float*)alloc(16 * 256 * 4);
  _Float16* wInT = (_Float16*)alloc((size_t)4 * 768 * 256 * 2);
  _Float16* wOutT = (_Float16*)alloc((size_t)4 * 256 * 256 * 2);
  _Float16* w1T = (_Float16*)alloc((size_t)32 * 256 * 256 * 2);
  _Float16* w2T = (_Float16*)alloc((size_t)32 * 256 * 256 * 2);
  _Float16* B16 = (_Float16*)alloc((size_t)4 * 64 * 256 * 2);
  _Float16* C16 = (_Float16*)alloc((size_t)4 * 256 * 64 * 2);
  _Float16* D16 = (_Float16*)alloc((size_t)4 * 256 * 256 * 2);
  _Float16* lmT = (_Float16*)alloc((size_t)32000 * 256 * 2);

  init_kernel<<<1, 32, 0, stream>>>(stats);

  // weight conversion / transposition to f16 [N][K]
  cvtT_kernel<<<dim3((768 * 256 + 255) / 256, 1, 4), 256, 0, stream>>>(attn_in_w, wInT, 256, 768);
  cvtT_kernel<<<dim3((256 * 256 + 255) / 256, 1, 4), 256, 0, stream>>>(attn_out_w, wOutT, 256, 256);
  cvtT_kernel<<<dim3((256 * 256 + 255) / 256, 1, 32), 256, 0, stream>>>(moe_w1, w1T, 256, 256);
  cvtT_kernel<<<dim3((256 * 256 + 255) / 256, 1, 32), 256, 0, stream>>>(moe_w2, w2T, 256, 256);
  cvtC_kernel<<<(4 * 64 * 256 + 255) / 256, 256, 0, stream>>>(ssm_Bm, B16, 4 * 64 * 256);
  cvtC_kernel<<<(4 * 256 * 64 + 255) / 256, 256, 0, stream>>>(ssm_Cm, C16, 4 * 256 * 64);
  cvtC_kernel<<<(4 * 256 * 256 + 255) / 256, 256, 0, stream>>>(ssm_Dm, D16, 4 * 256 * 256);
  cvtT_kernel<<<dim3((32000 * 256 + 255) / 256, 1, 1), 256, 0, stream>>>(lm_w, lmT, 256, 32000);

  embed_kernel<<<NTOK, 256, 0, stream>>>(tokens, tok_emb, pos_emb, x, xh);

  for (int i = 0; i < 4; ++i) {
    gemm_kernel<0, false, false, true, 256><<<dim3(6, 128), 256, 0, stream>>>(
        xh, wInT + (size_t)i * 768 * 256, attn_in_b + i * 768, qkvH,
        NTOK, 768, nullptr, 0, 0);
    attention_kernel<<<128, 256, 0, stream>>>(qkvH, aBufH);
    gemm_kernel<0, false, false, false, 256><<<dim3(2, 128), 256, 0, stream>>>(
        aBufH, wOutT + (size_t)i * 256 * 256, attn_out_b + i * 256, pBuf,
        NTOK, 256, nullptr, 0, 0);
    residual_ln_kernel<<<NTOK, 256, 0, stream>>>(x, xh, pBuf, ln1_g + i * 256, ln1_b + i * 256);
    gate_kernel<<<8, 256, 0, stream>>>(x, gate_w + i * 256 * 8, gate_b + i * 8,
                                       gateBuf, stats, stats + 1);
    for (int e = 0; e < 8; ++e) {
      const size_t wo = ((size_t)i * 8 + e) * 256 * 256;
      gemm_kernel<1, false, false, true, 256><<<dim3(2, 128), 256, 0, stream>>>(
          xh, w1T + wo, moe_b1 + ((size_t)i * 8 + e) * 256, hBufH,
          NTOK, 256, nullptr, 0, 0);
      if (e == 0)
        gemm_kernel<0, false, true, false, 256><<<dim3(2, 128), 256, 0, stream>>>(
            hBufH, w2T + wo, moe_b2 + ((size_t)i * 8 + e) * 256, moeBuf,
            NTOK, 256, gateBuf, 8, e);
      else
        gemm_kernel<0, true, true, false, 256><<<dim3(2, 128), 256, 0, stream>>>(
            hBufH, w2T + wo, moe_b2 + ((size_t)i * 8 + e) * 256, moeBuf,
            NTOK, 256, gateBuf, 8, e);
    }
    residual_ln_kernel<<<NTOK, 256, 0, stream>>>(x, xh, moeBuf, ln2_g + i * 256, ln2_b + i * 256);
    // SSM: xB projection -> recurrence -> hs@C^T + x@D^T
    gemm_kernel<0, false, false, false, 256><<<dim3(1, 128), 256, 0, stream>>>(
        xh, B16 + (size_t)i * 64 * 256, nullptr, xBBuf, NTOK, 64, nullptr, 0, 0);
    ssm_kernel<<<16, 64, 0, stream>>>(ssm_A + (size_t)i * 64 * 64, ssm_h0 + i * 64, xBBuf, hsH);
    gemm_kernel<0, false, false, false, 64><<<dim3(2, 128), 256, 0, stream>>>(
        hsH, C16 + (size_t)i * 256 * 64, nullptr, tmpBuf, NTOK, 256, nullptr, 0, 0);
    gemm_kernel<0, true, false, false, 256><<<dim3(2, 128), 256, 0, stream>>>(
        xh, D16 + (size_t)i * 256 * 256, nullptr, tmpBuf, NTOK, 256, nullptr, 0, 0);
    film_kernel<<<16, 256, 0, stream>>>(axw, film_gw + (size_t)i * 12 * 256, film_gb + i * 256,
                                        film_bw + (size_t)i * 12 * 256, film_bb + i * 256,
                                        gammaB, betaB);
    ssm_film_ln_kernel<<<NTOK, 256, 0, stream>>>(x, xh, tmpBuf, gammaB, betaB,
                                                 ln3_g + i * 256, ln3_b + i * 256);
  }

  pool_kernel<<<16, 256, 0, stream>>>(x, tokens, pooledB);
  ah1_kernel<<<16, 256, 0, stream>>>(pooledB, ah_w1, ah_b1, tB);
  ah2_kernel<<<16, 32, 0, stream>>>(tB, ah_w2, ah_b2, out);
  // LM head: 2048 x 32000, K=256
  gemm_kernel<0, false, false, false, 256><<<dim3(250, 128), 256, 0, stream>>>(
      xh, lmT, lm_b, out + 192, NTOK, 32000, nullptr, 0, 0);
  fin_kernel<<<1, 32, 0, stream>>>(stats, out + 192 + (size_t)NTOK * 32000,
                                   out + 192 + (size_t)NTOK * 32000 + 1);
}
